// AttentionBlock_12163347383129
// MI455X (gfx1250) — matrix operand order repaired and checked by reading, not compiled, not run on hardware
//
#include <hip/hip_runtime.h>

// ---------------- problem constants (fixed by reference) ----------------
#define BB   4
#define CC   256
#define TT   4096           // H*W
#define NH   4
#define CH   64             // channels per head
#define NG   32             // groups

typedef _Float16 h8  __attribute__((ext_vector_type(8)));
typedef _Float16 h16 __attribute__((ext_vector_type(16)));
typedef float    f8  __attribute__((ext_vector_type(8)));

#define DEVI static __device__ __forceinline__

// ---- WMMA fragment helpers (CDNA5 wave32 layouts, 16x16x32 f16) ----
// A (16x32, MxK): lane m=l&15, hi=l>>4; halves 0..7 -> K=8*hi..8*hi+7,
//                 halves 8..15 -> K=16+8*hi..23+8*hi.   (ISA 7.12.2)
DEVI h16 frag_a(const _Float16* base, int lda) {
  const int lane = threadIdx.x & 31;
  const int m = lane & 15, hi = lane >> 4;
  const _Float16* row = base + m * lda + 8 * hi;
  h8 lo = *(const h8*)(row);
  h8 hv = *(const h8*)(row + 16);
  h16 r;
#pragma unroll
  for (int i = 0; i < 8; ++i) { r[i] = lo[i]; r[i + 8] = hv[i]; }
  return r;
}
// B (32x16, KxN) from n-major storage Bt[n][k]: lane n=l&15, hi=l>>4;
// halves 0..7 -> K=8*hi..8*hi+7, halves 8..15 -> K=16+8*hi..23+8*hi (same as A).
DEVI h16 frag_b(const _Float16* base, int ldb) {
  const int lane = threadIdx.x & 31;
  const int n = lane & 15, hi = lane >> 4;
  const _Float16* p = base + n * ldb + 8 * hi;
  h8 lo = *(const h8*)(p);
  h8 hv = *(const h8*)(p + 16);
  h16 r;
#pragma unroll
  for (int i = 0; i < 8; ++i) { r[i] = lo[i]; r[i + 8] = hv[i]; }
  return r;
}
DEVI f8 wmma_f16(h16 a, h16 b, f8 c) {
  return __builtin_amdgcn_wmma_f32_16x16x32_f16(false, a, false, b, (short)0, c,
                                                false, false);
}
DEVI void cp32(_Float16* dst, const _Float16* src) {   // 32 halves via 16B ops
  *(h8*)(dst +  0) = *(const h8*)(src +  0);
  *(h8*)(dst +  8) = *(const h8*)(src +  8);
  *(h8*)(dst + 16) = *(const h8*)(src + 16);
  *(h8*)(dst + 24) = *(const h8*)(src + 24);
}
DEVI void cp16(_Float16* dst, const _Float16* src) {
  *(h8*)(dst + 0) = *(const h8*)(src + 0);
  *(h8*)(dst + 8) = *(const h8*)(src + 8);
}

// ---- CDNA5 async global->LDS (ASYNCcnt) -------------------------------
// GVS mode: mem_addr = SGPR_base + VGPR_off ; LDS addr = VGPR (byte offset,
// low 32 bits of the generic shared pointer: LDS aperture base has zero lo32).
DEVI unsigned lds_off(const _Float16* p) {
  return (unsigned)(unsigned long long)p;
}
DEVI void async_b128(unsigned lds, const _Float16* base, int voff_bytes) {
  asm volatile("global_load_async_to_lds_b128 %0, %1, %2"
               :: "v"(lds), "v"(voff_bytes), "s"(base)
               : "memory");
}
DEVI void async_wait8()  { asm volatile("s_wait_asynccnt 0x8" ::: "memory"); }
DEVI void async_wait0()  { asm volatile("s_wait_asynccnt 0x0" ::: "memory"); }

// ---------------- K0: fp32 -> f16 weight conversion ----------------
__global__ void cvt_kernel(const float* __restrict__ wq, const float* __restrict__ wp,
                           _Float16* __restrict__ wqh, _Float16* __restrict__ wph) {
  int i = blockIdx.x * 256 + threadIdx.x;
  if (i < 3 * CC * CC) wqh[i] = (_Float16)wq[i];
  if (i < CC * CC)     wph[i] = (_Float16)wp[i];
}

// ---------------- K1: GroupNorm -> h_t [B][T][C] f16 ----------------
__global__ void gn_kernel(const float* __restrict__ x, const float* __restrict__ gamma,
                          const float* __restrict__ beta, _Float16* __restrict__ ht) {
  const int g = blockIdx.x, b = blockIdx.y, tid = threadIdx.x;   // 32 groups x 4
  const float* base = x + ((size_t)b * CC + g * 8) * TT;         // 8 ch * 4096
  float s = 0.f, sq = 0.f;
#pragma unroll 4
  for (int i = tid; i < 8 * TT; i += 256) {
    float v = base[i];
    s += v; sq += v * v;
  }
  __shared__ float rs[256], rq[256];
  rs[tid] = s; rq[tid] = sq;
  __syncthreads();
  for (int st = 128; st > 0; st >>= 1) {
    if (tid < st) { rs[tid] += rs[tid + st]; rq[tid] += rq[tid + st]; }
    __syncthreads();
  }
  const float inv_n = 1.f / (8.f * TT);
  const float mean = rs[0] * inv_n;
  const float var  = rq[0] * inv_n - mean * mean;
  const float rinv = rsqrtf(var + 1e-5f);
  for (int i = tid; i < 8 * TT; i += 256) {
    int c8 = i >> 12, t = i & (TT - 1);
    int ch = g * 8 + c8;
    float v = (base[i] - mean) * rinv * gamma[ch] + beta[ch];
    ht[((size_t)b * TT + t) * CC + ch] = (_Float16)v;
  }
}

// ---------------- K2: QKV GEMM (768x4096x256 per batch) ----------------
// A = w_qkv f16 [o][c] ; Bt = h_t [t][c] ; scatter into q_t/k_t [bh][t][ch], v [bh][ch][t]
__global__ void __launch_bounds__(128)
qkv_kernel(const _Float16* __restrict__ ht, const _Float16* __restrict__ wq,
           const float* __restrict__ bias, _Float16* __restrict__ qt,
           _Float16* __restrict__ kt, _Float16* __restrict__ vv) {
  const int mblk = blockIdx.x, nblk = blockIdx.y, b = blockIdx.z;
  const int tid = threadIdx.x, w = tid >> 5, lane = tid & 31;
  const int m_off = (w & 1) * 32, n_off = (w >> 1) * 32;
  const int hi = lane >> 4, n16 = lane & 15;

  __shared__ _Float16 As[64 * 40];     // 64x32 tile, padded stride 40
  __shared__ _Float16 Bs[64 * 40];
  f8 acc[4] = {};
  const _Float16* Ag = wq + (size_t)(mblk * 64) * CC;
  const _Float16* Bg = ht + ((size_t)b * TT + nblk * 64) * CC;

  for (int kk = 0; kk < CC; kk += 32) {
    __syncthreads();
    int row = tid >> 1, part = (tid & 1) * 16;
    cp16(&As[row * 40 + part], &Ag[(size_t)row * CC + kk + part]);
    cp16(&Bs[row * 40 + part], &Bg[(size_t)row * CC + kk + part]);
    if (kk + 32 < CC)
      __builtin_prefetch(&Bg[(size_t)row * CC + kk + 32 + part], 0, 3);
    __syncthreads();
    h16 a0 = frag_a(As + (m_off +  0) * 40, 40);
    h16 a1 = frag_a(As + (m_off + 16) * 40, 40);
    h16 b0 = frag_b(Bs + (n_off +  0) * 40, 40);
    h16 b1 = frag_b(Bs + (n_off + 16) * 40, 40);
    acc[0] = wmma_f16(a0, b0, acc[0]);
    acc[1] = wmma_f16(a0, b1, acc[1]);
    acc[2] = wmma_f16(a1, b0, acc[2]);
    acc[3] = wmma_f16(a1, b1, acc[3]);
  }
  const float scale = 0.35355339059327373f;   // 64^-0.25
#pragma unroll
  for (int mi = 0; mi < 2; ++mi)
#pragma unroll
    for (int ni = 0; ni < 2; ++ni) {
      f8 d = acc[mi * 2 + ni];
#pragma unroll
      for (int r = 0; r < 8; ++r) {
        int m = mblk * 64 + m_off + mi * 16 + r + 8 * hi;    // 0..767
        int t = nblk * 64 + n_off + ni * 16 + n16;           // 0..4095
        float val = d[r] + bias[m];
        int head = m / 192, rr = m % 192;
        int which = rr >> 6, c = rr & 63;
        int bh = b * NH + head;
        if (which == 0)
          qt[((size_t)bh * TT + t) * CH + c] = (_Float16)(val * scale);
        else if (which == 1)
          kt[((size_t)bh * TT + t) * CH + c] = (_Float16)(val * scale);
        else
          vv[((size_t)bh * CH + c) * TT + t] = (_Float16)val;
      }
    }
}

// ---------------- K3: flash attention per (bh, 64-query tile) ----------------
// Double-buffered K/V tiles filled with async global->LDS (ASYNCcnt) loads.
__global__ void __launch_bounds__(128)
attn_kernel(const _Float16* __restrict__ qt, const _Float16* __restrict__ kt,
            const _Float16* __restrict__ vv, _Float16* __restrict__ at) {
  const int tblk = blockIdx.x, head = blockIdx.y, b = blockIdx.z;
  const int bh = b * NH + head;
  const int tid = threadIdx.x, w = tid >> 5, lane = tid & 31;
  const int hi = lane >> 4, n16 = lane & 15;

  __shared__ _Float16 Qs[64 * 72];        // padded stride 72 (bank spread)
  __shared__ _Float16 Ks[2][64 * 72];     // double-buffered
  __shared__ _Float16 Vs[2][64 * 72];     // Vs[c][s]
  __shared__ _Float16 Ps[64 * 72];

  const _Float16* qsrc = qt + ((size_t)bh * TT + tblk * 64) * CH;
  {
    int row = tid >> 1, part = (tid & 1) * 32;
    cp32(&Qs[row * 72 + part], &qsrc[(size_t)row * CH + part]);
  }

  const _Float16* ks0 = kt + (size_t)bh * TT * CH;   // [t][64] contiguous
  const _Float16* vs0 = vv + (size_t)bh * CH * TT;   // [c][T]

  const int row = tid >> 1, part = (tid & 1) * 32;   // per-thread 64B of K and V
  const unsigned kb[2] = { lds_off(&Ks[0][row * 72 + part]),
                           lds_off(&Ks[1][row * 72 + part]) };
  const unsigned vb[2] = { lds_off(&Vs[0][row * 72 + part]),
                           lds_off(&Vs[1][row * 72 + part]) };

  // issue async fill of buffer `buf` for key-block starting at `sb`
  auto issue = [&](int buf, int sb) {
    int kgo = ((sb + row) * CH + part) * 2;          // bytes into k_t
    int vgo = (row * TT + sb + part) * 2;            // bytes into v
#pragma unroll
    for (int c4 = 0; c4 < 4; ++c4) {
      async_b128(kb[buf] + c4 * 16, ks0, kgo + c4 * 16);
      async_b128(vb[buf] + c4 * 16, vs0, vgo + c4 * 16);
    }
  };

  issue(0, 0);                                       // prologue

  float m_i[8], l_i[8];
#pragma unroll
  for (int r = 0; r < 8; ++r) { m_i[r] = -1e30f; l_i[r] = 0.f; }
  f8 o[4] = {};

  for (int sb = 0; sb < TT; sb += 64) {
    const int cur = (sb >> 6) & 1;
    if (sb + 64 < TT) {                              // overlap next tile's DMA
      issue(cur ^ 1, sb + 64);
      async_wait8();                                 // current 8 chunks done
    } else {
      async_wait0();
    }
    __syncthreads();                                 // all waves' fills visible

    const _Float16* Kc = Ks[cur];
    const _Float16* Vc = Vs[cur];

    // S = Q K^T, 16 query rows per wave, 64 key cols
    f8 s[4] = {};
#pragma unroll
    for (int kk = 0; kk < 64; kk += 32) {
      h16 a = frag_a(Qs + (w * 16) * 72 + kk, 72);
#pragma unroll
      for (int nf = 0; nf < 4; ++nf) {
        h16 bmat = frag_b(Kc + (nf * 16) * 72 + kk, 72);
        s[nf] = wmma_f16(a, bmat, s[nf]);
      }
    }

    // online softmax (fp32): rows m=r+8*hi live in a 16-lane half
#pragma unroll
    for (int r = 0; r < 8; ++r) {
      float cm = fmaxf(fmaxf(s[0][r], s[1][r]), fmaxf(s[2][r], s[3][r]));
      cm = fmaxf(cm, __shfl_xor(cm, 1));
      cm = fmaxf(cm, __shfl_xor(cm, 2));
      cm = fmaxf(cm, __shfl_xor(cm, 4));
      cm = fmaxf(cm, __shfl_xor(cm, 8));
      float nm = fmaxf(m_i[r], cm);
      float alpha = __expf(m_i[r] - nm);
      float ps = 0.f;
#pragma unroll
      for (int nf = 0; nf < 4; ++nf) {
        float p = __expf(s[nf][r] - nm);
        ps += p;
        Ps[(w * 16 + r + 8 * hi) * 72 + nf * 16 + n16] = (_Float16)p;
      }
      ps += __shfl_xor(ps, 1);
      ps += __shfl_xor(ps, 2);
      ps += __shfl_xor(ps, 4);
      ps += __shfl_xor(ps, 8);
      l_i[r] = l_i[r] * alpha + ps;
      m_i[r] = nm;
#pragma unroll
      for (int nf = 0; nf < 4; ++nf) o[nf][r] = o[nf][r] * alpha;
    }

    // O += P V^T   (A = Ps rows of this wave only; Bt = Vs[c][s])
#pragma unroll
    for (int kk = 0; kk < 64; kk += 32) {
      h16 a = frag_a(Ps + (w * 16) * 72 + kk, 72);
#pragma unroll
      for (int nf = 0; nf < 4; ++nf) {
        h16 bmat = frag_b(Vc + (nf * 16) * 72 + kk, 72);
        o[nf] = wmma_f16(a, bmat, o[nf]);
      }
    }
    __syncthreads();   // all waves done with buf `cur` before it is refilled
  }

  // a_t[b][t][head*64 + c] = O / l
#pragma unroll
  for (int nf = 0; nf < 4; ++nf)
#pragma unroll
    for (int r = 0; r < 8; ++r) {
      int t = tblk * 64 + w * 16 + r + 8 * hi;
      int c = head * CH + nf * 16 + n16;
      at[((size_t)b * TT + t) * CC + c] = (_Float16)(o[nf][r] / l_i[r]);
    }
}

// ---------------- K4: proj GEMM + residual + bias -> fp32 out ----------------
__global__ void __launch_bounds__(128)
proj_kernel(const _Float16* __restrict__ at, const _Float16* __restrict__ wp,
            const float* __restrict__ x, const float* __restrict__ pbias,
            float* __restrict__ out) {
  const int mblk = blockIdx.x, nblk = blockIdx.y, b = blockIdx.z;
  const int tid = threadIdx.x, w = tid >> 5, lane = tid & 31;
  const int m_off = (w & 1) * 32, n_off = (w >> 1) * 32;
  const int hi = lane >> 4, n16 = lane & 15;

  __shared__ _Float16 As[64 * 40];
  __shared__ _Float16 Bs[64 * 40];
  f8 acc[4] = {};
  const _Float16* Ag = wp + (size_t)(mblk * 64) * CC;
  const _Float16* Bg = at + ((size_t)b * TT + nblk * 64) * CC;

  for (int kk = 0; kk < CC; kk += 32) {
    __syncthreads();
    int row = tid >> 1, part = (tid & 1) * 16;
    cp16(&As[row * 40 + part], &Ag[(size_t)row * CC + kk + part]);
    cp16(&Bs[row * 40 + part], &Bg[(size_t)row * CC + kk + part]);
    __syncthreads();
    h16 a0 = frag_a(As + (m_off +  0) * 40, 40);
    h16 a1 = frag_a(As + (m_off + 16) * 40, 40);
    h16 b0 = frag_b(Bs + (n_off +  0) * 40, 40);
    h16 b1 = frag_b(Bs + (n_off + 16) * 40, 40);
    acc[0] = wmma_f16(a0, b0, acc[0]);
    acc[1] = wmma_f16(a0, b1, acc[1]);
    acc[2] = wmma_f16(a1, b0, acc[2]);
    acc[3] = wmma_f16(a1, b1, acc[3]);
  }
#pragma unroll
  for (int mi = 0; mi < 2; ++mi)
#pragma unroll
    for (int ni = 0; ni < 2; ++ni) {
      f8 d = acc[mi * 2 + ni];
#pragma unroll
      for (int r = 0; r < 8; ++r) {
        int m = mblk * 64 + m_off + mi * 16 + r + 8 * hi;   // 0..255
        int t = nblk * 64 + n_off + ni * 16 + n16;
        size_t idx = ((size_t)b * CC + m) * TT + t;
        out[idx] = x[idx] + d[r] + pbias[m];
      }
    }
}

// ---------------- launch ----------------
extern "C" void kernel_launch(void* const* d_in, const int* in_sizes, int n_in,
                              void* d_out, int out_size, void* d_ws, size_t ws_size,
                              hipStream_t stream) {
  const float* x         = (const float*)d_in[0];
  const float* gamma     = (const float*)d_in[1];
  const float* beta      = (const float*)d_in[2];
  const float* w_qkv     = (const float*)d_in[3];
  const float* qkv_bias  = (const float*)d_in[4];
  const float* w_proj    = (const float*)d_in[5];
  const float* proj_bias = (const float*)d_in[6];
  float* out = (float*)d_out;

  char* ws = (char*)d_ws;
  const size_t MB = 1024 * 1024;
  _Float16* ht  = (_Float16*)(ws + 0 * MB);   // [B][T][C]        8 MB
  _Float16* qt  = (_Float16*)(ws + 8 * MB);   // [B*NH][T][CH]    8 MB
  _Float16* kt  = (_Float16*)(ws + 16 * MB);  // [B*NH][T][CH]    8 MB
  _Float16* vv  = (_Float16*)(ws + 24 * MB);  // [B*NH][CH][T]    8 MB
  _Float16* at  = (_Float16*)(ws + 32 * MB);  // [B][T][C]        8 MB
  _Float16* wqh = (_Float16*)(ws + 40 * MB);  // 768x256 f16
  _Float16* wph = (_Float16*)(ws + 41 * MB);  // 256x256 f16

  cvt_kernel<<<(3 * CC * CC + 255) / 256, 256, 0, stream>>>(w_qkv, w_proj, wqh, wph);
  gn_kernel<<<dim3(NG, BB), 256, 0, stream>>>(x, gamma, beta, ht);
  qkv_kernel<<<dim3(12, TT / 64, BB), 128, 0, stream>>>(ht, wqh, qkv_bias, qt, kt, vv);
  attn_kernel<<<dim3(TT / 64, NH, BB), 128, 0, stream>>>(qt, kt, vv, at);
  proj_kernel<<<dim3(CC / 64, TT / 64, BB), 128, 0, stream>>>(at, wph, x, proj_bias, out);
}
